// Gaussian_NN_Htanh_71279277245132
// MI455X (gfx1250) — matrix operand order repaired and checked by reading, not compiled, not run on hardware
//
#include <hip/hip_runtime.h>

// ---------------------------------------------------------------------------
// Gaussian NN (reparameterized 2-layer MLP) + Hardtanh, J=8 ensemble.
//   h   = clip(x @ (muW1 + eW1[j]*sigW1) + b1[j], -1, 1)   [J,N,H]  (bf16, ws)
//   out = h @ (muW2 + eW2[j]*sigW2) + b2[j]                [J,N,Y]  (f32)
// bf16 WMMA (V_WMMA_F32_16X16X32_BF16) w/ f32 accumulation. ~51.5 GFLOP over
// ~145 MB mandatory HBM traffic -> bandwidth-bound at 23.3 TB/s (~6 us); the
// bf16 matrix path keeps matrix-core time at/below that. All global loads are
// b128; h-tile copy in kernel 2 uses GLOBAL_LOAD_ASYNC_TO_LDS_B128.
// ---------------------------------------------------------------------------

#define XDIM   1024
#define HDIM   2048
#define YDIM   512
#define NBATCH 1024
#define JENS   8

#define BM 128
#define BN 128
#define BK 32
#define LDS_STRIDE 40   // bf16 elems per LDS row: 80 B (multiple of 16 B)

typedef __attribute__((ext_vector_type(16))) __bf16 bf16x16;
typedef __attribute__((ext_vector_type(8)))  float  f32x8;
typedef __attribute__((ext_vector_type(4)))  int    v4i;

union Frag16 {
    bf16x16 v;
    struct { uint4 lo; uint4 hi; } q;
};

union BF4 {            // 4 bf16 <-> 8-byte store
    __bf16 h[4];
    uint2  u;
};

#if __has_builtin(__builtin_amdgcn_global_load_async_to_lds_b128)
#define HAVE_ASYNC_LDS 1
typedef __attribute__((address_space(1))) v4i g_v4i;   // global int4
typedef __attribute__((address_space(3))) v4i l_v4i;   // LDS int4

__device__ __forceinline__ void async_copy_b128(const void* g, void* l) {
    __builtin_amdgcn_global_load_async_to_lds_b128(
        (g_v4i*)g, (l_v4i*)l, 0, 0);
}
#endif

// ---------------------------------------------------------------------------
// Kernel 1: h_bf16[j] = clip(x @ W1[j] + b1[j]) with W1 built in registers.
// ---------------------------------------------------------------------------
__global__ __launch_bounds__(256)
void k_gemm1_htanh(const float* __restrict__ x,
                   const float* __restrict__ muW1,
                   const float* __restrict__ sigW1,
                   const float* __restrict__ eW1,
                   const float* __restrict__ mub1,
                   const float* __restrict__ sigb1,
                   const float* __restrict__ eb1,
                   __bf16*      __restrict__ hout)
{
    __shared__ __bf16 As[BM * LDS_STRIDE];   // [m][k]       (row-major)
    __shared__ __bf16 Bs[BN * LDS_STRIDE];   // [h_col][k]   (K-contiguous)

    const int j    = blockIdx.z;
    const int m0   = blockIdx.y * BM;
    const int h0   = blockIdx.x * BN;

    const int tid   = threadIdx.x;
    const int lane  = tid & 31;
    const int wave  = tid >> 5;
    const int wm    = wave >> 2;       // 0..1 : 64 rows each
    const int wn    = wave & 3;        // 0..3 : 32 cols each
    const int lmod  = lane & 15;
    const int lhalf = lane >> 4;

    const float* eW1j = eW1 + (size_t)j * XDIM * HDIM;

    f32x8 acc[4][2];
    #pragma unroll
    for (int mi = 0; mi < 4; ++mi)
        #pragma unroll
        for (int ni = 0; ni < 2; ++ni)
            #pragma unroll
            for (int r = 0; r < 8; ++r) acc[mi][ni][r] = 0.0f;

    for (int k0 = 0; k0 < XDIM; k0 += BK) {
        // ---- x tile [BM x BK]: b128 loads, packed b64 LDS stores ----------
        #pragma unroll
        for (int i = 0; i < 4; ++i) {
            const int idx = tid * 4 + i * 1024;
            const int r   = idx >> 5;      // /BK
            const int c   = idx & 31;
            const float4 v =
                *(const float4*)&x[(size_t)(m0 + r) * XDIM + (k0 + c)];
            BF4 p;
            p.h[0] = (__bf16)v.x; p.h[1] = (__bf16)v.y;
            p.h[2] = (__bf16)v.z; p.h[3] = (__bf16)v.w;
            *(uint2*)&As[r * LDS_STRIDE + c] = p.u;
        }
        // ---- W1 tile [BK x BN]: b128 loads, mu + e*sig, K-contiguous ------
        #pragma unroll
        for (int i = 0; i < 4; ++i) {
            const int idx = tid * 4 + i * 1024;
            const int kr  = idx >> 7;      // /BN
            const int hc  = idx & 127;
            const size_t g = (size_t)(k0 + kr) * HDIM + (h0 + hc);
            const float4 m = *(const float4*)&muW1[g];
            const float4 e = *(const float4*)&eW1j[g];
            const float4 s = *(const float4*)&sigW1[g];
            Bs[(hc + 0) * LDS_STRIDE + kr] = (__bf16)(m.x + e.x * s.x);
            Bs[(hc + 1) * LDS_STRIDE + kr] = (__bf16)(m.y + e.y * s.y);
            Bs[(hc + 2) * LDS_STRIDE + kr] = (__bf16)(m.z + e.z * s.z);
            Bs[(hc + 3) * LDS_STRIDE + kr] = (__bf16)(m.w + e.w * s.w);
            if (i == 0 && k0 + BK < XDIM) {
                __builtin_prefetch(eW1j + g + (size_t)BK * HDIM, 0, 0);
            }
        }
        __syncthreads();

        // ---- B fragments: lane holds col n0+lmod, K = lhalf*16 .. +15 -----
        Frag16 bf[2];
        #pragma unroll
        for (int ni = 0; ni < 2; ++ni) {
            const __bf16* bp =
                &Bs[(wn * 32 + ni * 16 + lmod) * LDS_STRIDE + lhalf * 16];
            bf[ni].q.lo = *(const uint4*)(bp);
            bf[ni].q.hi = *(const uint4*)(bp + 8);
        }
        // ---- A fragments + WMMA ------------------------------------------
        #pragma unroll
        for (int mi = 0; mi < 4; ++mi) {
            const __bf16* ap =
                &As[(wm * 64 + mi * 16 + lmod) * LDS_STRIDE + lhalf * 16];
            Frag16 af;
            af.q.lo = *(const uint4*)(ap);
            af.q.hi = *(const uint4*)(ap + 8);
            #pragma unroll
            for (int ni = 0; ni < 2; ++ni) {
                acc[mi][ni] = __builtin_amdgcn_wmma_f32_16x16x32_bf16(
                    false, af.v, false, bf[ni].v,
                    (short)0, acc[mi][ni], false, false);
            }
        }
        __syncthreads();
    }

    // ---- epilogue: + b1[j], hardtanh, bf16 store -------------------------
    #pragma unroll
    for (int ni = 0; ni < 2; ++ni) {
        const int h = h0 + wn * 32 + ni * 16 + lmod;
        const float bias = mub1[h] + eb1[(size_t)j * HDIM + h] * sigb1[h];
        #pragma unroll
        for (int mi = 0; mi < 4; ++mi) {
            #pragma unroll
            for (int r = 0; r < 8; ++r) {
                const int m = m0 + wm * 64 + mi * 16 + lhalf * 8 + r;
                float v = acc[mi][ni][r] + bias;
                v = fminf(1.0f, fmaxf(-1.0f, v));
                hout[((size_t)j * NBATCH + m) * HDIM + h] = (__bf16)v;
            }
        }
    }
}

// ---------------------------------------------------------------------------
// Kernel 2: out[j] = h_bf16[j] @ W2[j] + b2[j]   (f32 output)
// ---------------------------------------------------------------------------
__global__ __launch_bounds__(256)
void k_gemm2(const __bf16* __restrict__ h,
             const float*  __restrict__ muW2,
             const float*  __restrict__ sigW2,
             const float*  __restrict__ eW2,
             const float*  __restrict__ mub2,
             const float*  __restrict__ sigb2,
             const float*  __restrict__ eb2,
             float*        __restrict__ out)
{
    __shared__ __bf16 As[BM * LDS_STRIDE];
    __shared__ __bf16 Bs[BN * LDS_STRIDE];

    const int j    = blockIdx.z;
    const int m0   = blockIdx.y * BM;
    const int y0   = blockIdx.x * BN;

    const int tid   = threadIdx.x;
    const int lane  = tid & 31;
    const int wave  = tid >> 5;
    const int wm    = wave >> 2;
    const int wn    = wave & 3;
    const int lmod  = lane & 15;
    const int lhalf = lane >> 4;

    const __bf16* hj   = h   + (size_t)j * NBATCH * HDIM;
    const float*  eW2j = eW2 + (size_t)j * HDIM * YDIM;

    f32x8 acc[4][2];
    #pragma unroll
    for (int mi = 0; mi < 4; ++mi)
        #pragma unroll
        for (int ni = 0; ni < 2; ++ni)
            #pragma unroll
            for (int r = 0; r < 8; ++r) acc[mi][ni][r] = 0.0f;

    for (int k0 = 0; k0 < HDIM; k0 += BK) {
        // ---- h tile [BM x BK] bf16: async copy to LDS when available -----
        {
            const int r  = tid >> 1;
            const int ko = (tid & 1) * 16;
            const __bf16* gp = &hj[(size_t)(m0 + r) * HDIM + (k0 + ko)];
            __bf16* sp = &As[r * LDS_STRIDE + ko];
#ifdef HAVE_ASYNC_LDS
            async_copy_b128(gp,     sp);
            async_copy_b128(gp + 8, sp + 8);
#else
            const uint4 lo = *(const uint4*)(gp);
            const uint4 hi = *(const uint4*)(gp + 8);
            *(uint4*)(sp)     = lo;
            *(uint4*)(sp + 8) = hi;
#endif
        }
        // ---- W2 tile [BK x BN]: b128 loads, mu + e*sig, K-contiguous -----
        #pragma unroll
        for (int i = 0; i < 4; ++i) {
            const int idx = tid * 4 + i * 1024;
            const int kr  = idx >> 7;
            const int yc  = idx & 127;
            const size_t g = (size_t)(k0 + kr) * YDIM + (y0 + yc);
            const float4 m = *(const float4*)&muW2[g];
            const float4 e = *(const float4*)&eW2j[g];
            const float4 s = *(const float4*)&sigW2[g];
            Bs[(yc + 0) * LDS_STRIDE + kr] = (__bf16)(m.x + e.x * s.x);
            Bs[(yc + 1) * LDS_STRIDE + kr] = (__bf16)(m.y + e.y * s.y);
            Bs[(yc + 2) * LDS_STRIDE + kr] = (__bf16)(m.z + e.z * s.z);
            Bs[(yc + 3) * LDS_STRIDE + kr] = (__bf16)(m.w + e.w * s.w);
            if (i == 0 && k0 + BK < HDIM) {
                __builtin_prefetch(eW2j + g + (size_t)BK * YDIM, 0, 0);
            }
        }
#ifdef HAVE_ASYNC_LDS
#if __has_builtin(__builtin_amdgcn_s_wait_asynccnt)
        __builtin_amdgcn_s_wait_asynccnt(0);
#else
        asm volatile("s_wait_asynccnt 0x0" ::: "memory");
#endif
#endif
        __syncthreads();

        Frag16 bf[2];
        #pragma unroll
        for (int ni = 0; ni < 2; ++ni) {
            const __bf16* bp =
                &Bs[(wn * 32 + ni * 16 + lmod) * LDS_STRIDE + lhalf * 16];
            bf[ni].q.lo = *(const uint4*)(bp);
            bf[ni].q.hi = *(const uint4*)(bp + 8);
        }
        #pragma unroll
        for (int mi = 0; mi < 4; ++mi) {
            const __bf16* ap =
                &As[(wm * 64 + mi * 16 + lmod) * LDS_STRIDE + lhalf * 16];
            Frag16 af;
            af.q.lo = *(const uint4*)(ap);
            af.q.hi = *(const uint4*)(ap + 8);
            #pragma unroll
            for (int ni = 0; ni < 2; ++ni) {
                acc[mi][ni] = __builtin_amdgcn_wmma_f32_16x16x32_bf16(
                    false, af.v, false, bf[ni].v,
                    (short)0, acc[mi][ni], false, false);
            }
        }
        __syncthreads();
    }

    // ---- epilogue: + b2[j], f32 store ------------------------------------
    #pragma unroll
    for (int ni = 0; ni < 2; ++ni) {
        const int y = y0 + wn * 32 + ni * 16 + lmod;
        const float bias = mub2[y] + eb2[(size_t)j * YDIM + y] * sigb2[y];
        #pragma unroll
        for (int mi = 0; mi < 4; ++mi) {
            #pragma unroll
            for (int r = 0; r < 8; ++r) {
                const int m = m0 + wm * 64 + mi * 16 + lhalf * 8 + r;
                out[((size_t)j * NBATCH + m) * YDIM + y] = acc[mi][ni][r] + bias;
            }
        }
    }
}

// ---------------------------------------------------------------------------
// kernel_launch
// Inputs (setup_inputs order):
//  0:x 1:mu_W1 2:mu_b1 3:mu_W2 4:mu_b2 5:sig_W1 6:sig_b1 7:sig_W2 8:sig_b2
//  9:e_W1 10:e_b1 11:e_W2 12:e_b2
// Workspace: h as bf16 [J, N, H] = 8*1024*2048*2 = 33,554,432 bytes.
// ---------------------------------------------------------------------------
extern "C" void kernel_launch(void* const* d_in, const int* in_sizes, int n_in,
                              void* d_out, int out_size, void* d_ws, size_t ws_size,
                              hipStream_t stream)
{
    const float* x     = (const float*)d_in[0];
    const float* muW1  = (const float*)d_in[1];
    const float* mub1  = (const float*)d_in[2];
    const float* muW2  = (const float*)d_in[3];
    const float* mub2  = (const float*)d_in[4];
    const float* sigW1 = (const float*)d_in[5];
    const float* sigb1 = (const float*)d_in[6];
    const float* sigW2 = (const float*)d_in[7];
    const float* sigb2 = (const float*)d_in[8];
    const float* eW1   = (const float*)d_in[9];
    const float* eb1   = (const float*)d_in[10];
    const float* eW2   = (const float*)d_in[11];
    const float* eb2   = (const float*)d_in[12];

    __bf16* h_ws = (__bf16*)d_ws;            // [J, N, H] bf16
    float*  outp = (float*)d_out;            // [J, N, Y] f32

    dim3 grid1(HDIM / BN, NBATCH / BM, JENS);   // 16 x 8 x 8
    dim3 grid2(YDIM / BN, NBATCH / BM, JENS);   //  4 x 8 x 8
    dim3 block(256);

    k_gemm1_htanh<<<grid1, block, 0, stream>>>(x, muW1, sigW1, eW1,
                                               mub1, sigb1, eb1, h_ws);
    k_gemm2<<<grid2, block, 0, stream>>>(h_ws, muW2, sigW2, eW2,
                                         mub2, sigb2, eb2, outp);
}
